// Detector_1305670058340
// MI455X (gfx1250) — matrix operand order repaired and checked by reading, not compiled, not run on hardware
//
#include <hip/hip_runtime.h>
#include <stdint.h>

#define QN 4
#define RFN 32
#define NSAMP 4
#define CN 18
#define LVLS 3
#define CONVN 6
#define HH 64
#define WWD 64
#define HWSZ 4096
#define DD 64
#define NPIX 65536   // QN*NSAMP*H*W
#define NTILE 4096   // NPIX/16

typedef __attribute__((ext_vector_type(16))) _Float16 v16h;
typedef __attribute__((ext_vector_type(8)))  _Float16 v8h;
typedef __attribute__((ext_vector_type(8)))  float    v8f;

__device__ __forceinline__ v8f wmma16(v16h a, v16h b, v8f c) {
  return __builtin_amdgcn_wmma_f32_16x16x32_f16(false, a, false, b, (short)0, c, false, false);
}

// ---------------- zero-fill (packed B buffers need true zero padding) ------
__global__ void k_zero(uint32_t* __restrict__ p, int n) {
  int i = blockIdx.x * 256 + threadIdx.x;
  if (i < n) p[i] = 0u;
}

// ---------------- Stage 1: instance-norm statistics per (q,r,s,c) plane ----
__global__ void k_stats(const float* __restrict__ corr, float* __restrict__ stats) {
  int plane = blockIdx.x;                       // 9216 planes
  const float* p = corr + (size_t)plane * HWSZ;
  float s = 0.f, s2 = 0.f;
  for (int i = threadIdx.x; i < HWSZ; i += 256) { float v = p[i]; s += v; s2 += v * v; }
  __shared__ float sh1[256], sh2[256];
  sh1[threadIdx.x] = s; sh2[threadIdx.x] = s2; __syncthreads();
  for (int st = 128; st > 0; st >>= 1) {
    if (threadIdx.x < st) { sh1[threadIdx.x] += sh1[threadIdx.x + st]; sh2[threadIdx.x] += sh2[threadIdx.x + st]; }
    __syncthreads();
  }
  if (threadIdx.x == 0) {
    float m   = sh1[0] * (1.f / HWSZ);
    float var = sh2[0] * (1.f / HWSZ) - m * m;
    stats[2 * plane]     = m;
    stats[2 * plane + 1] = rsqrtf(var + 1e-5f);
  }
}

// ---------------- Stage 2: level means of normalized corr ------------------
__global__ void k_lvl(const float* __restrict__ corr, const float* __restrict__ stats,
                      float* __restrict__ lvl) {
  size_t t = (size_t)blockIdx.x * 256 + threadIdx.x;   // 6,291,456 total
  int hw = (int)(t & 4095);
  size_t u = t >> 12;
  int l = (int)(u % 3); u /= 3;
  int s = (int)(u & 3); u >>= 2;
  int r = (int)(u & 31);
  int q = (int)(u >> 5);
  int pbase = ((q * RFN + r) * NSAMP + s) * CN + l * CONVN;
  float acc = 0.f;
#pragma unroll
  for (int j = 0; j < CONVN; ++j) {
    int plane = pbase + j;
    acc += (corr[(size_t)plane * HWSZ + hw] - stats[2 * plane]) * stats[2 * plane + 1];
  }
  lvl[t] = acc * (1.f / 6.f);
}

// ---------------- Stage 3: top-4 over rfn per (q,s,h,w) --------------------
__global__ void k_topk(const float* __restrict__ lvl, int* __restrict__ topi) {
  int pixel = blockIdx.x * 256 + threadIdx.x;
  int q = pixel >> 14; int s = (pixel >> 12) & 3; int hw = pixel & 4095;
  float v0 = -3e38f, v1 = -3e38f, v2 = -3e38f, v3 = -3e38f;
  int   i0 = 0, i1 = 0, i2 = 0, i3 = 0;
  for (int r = 0; r < RFN; ++r) {
    size_t base = (size_t)(((q * RFN + r) * NSAMP + s) * LVLS) * HWSZ + hw;
    float v = (lvl[base] + lvl[base + HWSZ] + lvl[base + 2 * HWSZ]) * (1.f / 3.f);
    if      (v > v0) { v3=v2;i3=i2; v2=v1;i2=i1; v1=v0;i1=i0; v0=v;i0=r; }
    else if (v > v1) { v3=v2;i3=i2; v2=v1;i2=i1; v1=v;i1=r; }
    else if (v > v2) { v3=v2;i3=i2; v2=v;i2=r; }
    else if (v > v3) { v3=v;i3=r; }
  }
  topi[pixel*4+0]=i0; topi[pixel*4+1]=i1; topi[pixel*4+2]=i2; topi[pixel*4+3]=i3;
}

// ---------------- Stage 4: KNN gather -> packed B-fragment operands --------
// B frag layout: frag (nt,ch) is 512 halves; element (lane,e): n=lane&15,
// k = 32*ch + 16*(lane>>4) + e.  We write: lane = ((kc&16)?16:0)|n, e = kc&15.
__global__ void k_gather(const float* __restrict__ corr, const float* __restrict__ stats,
                         const float* __restrict__ lvl, const int* __restrict__ topi,
                         _Float16* __restrict__ xscale_p,   // [NTILE][3][512]
                         _Float16* __restrict__ xscore_p) { // [NTILE][1][512]
  int t = blockIdx.x * 256 + threadIdx.x;              // kk*65536 + pixel
  int pixel = t & (NPIX - 1);
  int kk = t >> 16;
  int q = pixel >> 14; int s = (pixel >> 12) & 3; int hw = pixel & 4095;
  int n  = pixel & 15;
  int nt = pixel >> 4;
  int r = topi[pixel * 4 + kk];
  int pbase = ((q * RFN + r) * NSAMP + s) * CN;
#pragma unroll
  for (int c = 0; c < CN; ++c) {
    int plane = pbase + c;
    float v = (corr[(size_t)plane * HWSZ + hw] - stats[2 * plane]) * stats[2 * plane + 1];
    int k  = kk * CN + c;               // 0..71
    int ch = k >> 5, kc = k & 31;
    int lane = ((kc & 16) ? 16 : 0) | n;
    xscale_p[((size_t)(nt * 3 + ch) << 9) + lane * 16 + (kc & 15)] = (_Float16)v;
  }
  size_t lbase = (size_t)(((q * RFN + r) * NSAMP + s) * LVLS) * HWSZ + hw;
#pragma unroll
  for (int l = 0; l < LVLS; ++l) {
    int k = kk * LVLS + l;              // 0..11 (rest stays zero-filled)
    xscore_p[((size_t)nt << 9) + n * 16 + k] = (_Float16)lvl[lbase + (size_t)l * HWSZ];
  }
}

// ---------------- A-fragment packer (weights -> WMMA register layout) ------
// dst[((mt*ntaps+tap)*nch+ch)*512 + lane*16 + e]; A layout: K=e+16*hi
// src layout: (o, cin, taps) i.e. src[(o*cin + k)*ntaps + tap]
__global__ void k_packA(const float* __restrict__ src, _Float16* __restrict__ dst,
                        int nmt, int ntaps, int nch, int cout, int cin) {
  int t = blockIdx.x * 256 + threadIdx.x;
  int total = nmt * ntaps * nch * 512;
  if (t >= total) return;
  int e = t & 15;
  int lane = (t >> 4) & 31;
  int rest = t >> 9;
  int ch = rest % nch; rest /= nch;
  int tap = rest % ntaps;
  int mt = rest / ntaps;
  int o  = mt * 16 + (lane & 15);
  int kc = e + 16 * (lane >> 4);
  int k  = 32 * ch + kc;
  float v = (o < cout && k < cin) ? src[(o * cin + k) * ntaps + tap] : 0.f;
  dst[t] = (_Float16)v;
}

// ---------------- Stage 5: fused 1x1x1 two-layer GEMM chain (WMMA) ---------
// Xp: packed B frags [NTILE][nch1][512]; w1p: [4][nch1][512]; w2p: [4][2][512]
// out: pixel-major f32 [pixel][64]
__global__ void k_mlp(const _Float16* __restrict__ Xp, int nch1,
                      const _Float16* __restrict__ w1p, const float* __restrict__ b1,
                      const _Float16* __restrict__ w2p, const float* __restrict__ b2,
                      float* __restrict__ out) {
  int lane = threadIdx.x & 31;
  int wv   = threadIdx.x >> 5;          // m-tile 0..3
  int hi = lane >> 4, n = lane & 15;
  int nt = blockIdx.x;
  v8f acc = {};
  for (int ch = 0; ch < nch1; ++ch) {
    v16h a = *(const v16h*)(w1p + (((size_t)wv * nch1 + ch) << 9) + lane * 16);
    v16h b = *(const v16h*)(Xp  + (((size_t)nt * nch1 + ch) << 9) + lane * 16);
    acc = wmma16(a, b, acc);
  }
  // layer-1 epilogue: bias+relu, scatter into LDS in layer-2 B-frag layout
  __shared__ __attribute__((aligned(32))) _Float16 l2b[2 * 512];
#pragma unroll
  for (int r2 = 0; r2 < 8; ++r2) {
    int o = wv * 16 + r2 + 8 * hi;      // C frag: M = r2 + 8*hi
    float v = acc[r2] + b1[o];
    v = v > 0.f ? v : 0.f;
    int ch2 = o >> 5, kc = o & 31;
    int tl = ((kc & 16) ? 16 : 0) | n;
    l2b[(ch2 << 9) + tl * 16 + (kc & 15)] = (_Float16)v;
  }
  __syncthreads();
  v8f acc2 = {};
#pragma unroll
  for (int ch = 0; ch < 2; ++ch) {
    v16h a = *(const v16h*)(w2p + (((size_t)wv * 2 + ch) << 9) + lane * 16);
    v16h b = *(const v16h*)(l2b + (ch << 9) + lane * 16);
    acc2 = wmma16(a, b, acc2);
  }
  v8f res;
#pragma unroll
  for (int r2 = 0; r2 < 8; ++r2) res[r2] = acc2[r2] + b2[wv * 16 + r2 + 8 * hi];
  int pixel = nt * 16 + n;
  *(v8f*)(out + ((size_t)pixel << 6) + wv * 16 + 8 * hi) = res;
}

// ---------------- Stage 6: max over s, emit pixel-major f16 head input -----
__global__ void k_maxs(const float* __restrict__ src, _Float16* __restrict__ dst) {
  int t = blockIdx.x * 256 + threadIdx.x;              // 1,048,576 total
  int o  = t & 63;
  int hw = (t >> 6) & 4095;
  int q  = t >> 18;
  float mx = -3e38f;
#pragma unroll
  for (int s = 0; s < NSAMP; ++s) {
    float v = src[(((size_t)(q * NSAMP + s) << 12) + hw) * 64 + o];
    mx = mx > v ? mx : v;
  }
  dst[(((size_t)q << 12) + hw) * 64 + o] = (_Float16)mx;
}

// ---------------- Stage 7: 3x3 conv via WMMA, LDS-staged B fragments -------
// in: pixel-major f16 [q][h][w][64]; wp: packed A frags [nmt][9][2][512]
// outh: pixel-major f16 [q][h][w][64] (cout==64); outf: planar f32 [q][cout][hw]
__global__ void k_conv3(const _Float16* __restrict__ in, const _Float16* __restrict__ wp,
                        const float* __restrict__ bias, int cout, int dorelu,
                        _Float16* __restrict__ outh, float* __restrict__ outf) {
  int lane = threadIdx.x & 31;
  int mt   = threadIdx.x >> 5;
  int bid = blockIdx.x;                 // 1024 = 4(wt) * 64(h) * 4(q)
  int wt = bid & 3;
  int h  = (bid >> 2) & 63;
  int q  = bid >> 8;
  int hi = lane >> 4, n = lane & 15;
  int w0 = wt * 16;

  // phase 1: build all 18 B fragments (9 taps x 2 K-chunks) cooperatively
  __shared__ __attribute__((aligned(32))) _Float16 bfr[18 * 512];
  for (int row = threadIdx.x; row < 18 * 32; row += blockDim.x) {
    int frag = row >> 5;                // tap*2 + ch
    int lr   = row & 31;
    int tap = frag >> 1, ch = frag & 1;
    int dy = tap / 3 - 1, dx = tap % 3 - 1;
    int nn = lr & 15, hb = lr >> 4;
    int hh = h + dy;
    int wx = w0 + nn + dx;
    int k0 = 32 * ch + 16 * hb;
    v16h v = {};
    if (hh >= 0 && hh < HH && wx >= 0 && wx < WWD)
      v = *(const v16h*)(in + ((((size_t)q * HH + hh) * WWD + wx) << 6) + k0);
    *(v16h*)(bfr + (frag << 9) + lr * 16) = v;
  }
  __syncthreads();

  // phase 2: 18 WMMAs per wave, contiguous A (global) + B (LDS) vector loads
  v8f acc = {};
#pragma unroll
  for (int frag = 0; frag < 18; ++frag) {
    v16h a = *(const v16h*)(wp + (((size_t)mt * 18 + frag) << 9) + lane * 16);
    v16h b = *(const v16h*)(bfr + (frag << 9) + lane * 16);
    acc = wmma16(a, b, acc);
  }

  if (outh) {                           // full-width layer: vector f16 store
    v8h hv;
#pragma unroll
    for (int r2 = 0; r2 < 8; ++r2) {
      int o = mt * 16 + r2 + 8 * hi;
      float v = acc[r2] + bias[o];
      if (dorelu) v = v > 0.f ? v : 0.f;
      hv[r2] = (_Float16)v;
    }
    *(v8h*)(outh + ((((size_t)q * HH + h) * WWD + w0 + n) << 6) + mt * 16 + 8 * hi) = hv;
  }
  if (outf) {                           // narrow head output: planar f32
#pragma unroll
    for (int r2 = 0; r2 < 8; ++r2) {
      int o = mt * 16 + r2 + 8 * hi;
      if (o < cout) {
        float v = acc[r2] + bias[o];
        outf[(((size_t)q * cout + o) << 12) + h * WWD + w0 + n] = v;
      }
    }
  }
}

// ---------------- Stage 8: argmax + pos/scale epilogue ---------------------
__global__ void k_final(const float* __restrict__ scores, const float* __restrict__ scales,
                        const float* __restrict__ offsets, float* __restrict__ pos,
                        float* __restrict__ scl) {
  int q = blockIdx.x;
  __shared__ float bv[256]; __shared__ int bi[256];
  float best = -3e38f; int bidx = 0;
  for (int i = threadIdx.x; i < HWSZ; i += 256) {
    float v = scores[q * HWSZ + i];
    if (v > best) { best = v; bidx = i; }
  }
  bv[threadIdx.x] = best; bi[threadIdx.x] = bidx; __syncthreads();
  for (int st = 128; st > 0; st >>= 1) {
    if (threadIdx.x < st) {
      if (bv[threadIdx.x + st] > bv[threadIdx.x] ||
          (bv[threadIdx.x + st] == bv[threadIdx.x] && bi[threadIdx.x + st] < bi[threadIdx.x])) {
        bv[threadIdx.x] = bv[threadIdx.x + st]; bi[threadIdx.x] = bi[threadIdx.x + st];
      }
    }
    __syncthreads();
  }
  if (threadIdx.x == 0) {
    int idx = bi[0]; int sy = idx >> 6; int sx = idx & 63;
    float ox = offsets[(q * 2 + 0) * HWSZ + idx];
    float oy = offsets[(q * 2 + 1) * HWSZ + idx];
    pos[q * 2 + 0] = ((float)sx + ox + 0.5f) * 8.f - 0.5f;
    pos[q * 2 + 1] = ((float)sy + oy + 0.5f) * 8.f - 0.5f;
    scl[q] = exp2f(scales[q * HWSZ + idx]);
  }
}

// ===========================================================================
extern "C" void kernel_launch(void* const* d_in, const int* in_sizes, int n_in,
                              void* d_out, int out_size, void* d_ws, size_t ws_size,
                              hipStream_t stream) {
  (void)in_sizes; (void)n_in; (void)out_size; (void)ws_size;
  const float* corr = (const float*)d_in[0];

  // ---- carve workspace (~68 MB) ----
  char* ws = (char*)d_ws;
  size_t off = 0;
  auto carve = [&](size_t bytes) -> char* {
    char* p = ws + off; off += (bytes + 255) & ~(size_t)255; return p;
  };
  float*    stats    = (float*)carve((size_t)9216 * 2 * 4);
  float*    lvl      = (float*)carve((size_t)6291456 * 4);
  int*      topi     = (int*)carve((size_t)NPIX * 4 * 4);
  _Float16* xscale_p = (_Float16*)carve((size_t)NTILE * 3 * 512 * 2); // 12.6 MB
  _Float16* xscore_p = (_Float16*)carve((size_t)NTILE * 1 * 512 * 2); //  4.2 MB
  float*    out2     = (float*)carve((size_t)NPIX * DD * 4);          // 16 MB
  _Float16* xs_h     = (_Float16*)carve((size_t)QN * HWSZ * DD * 2);
  _Float16* xc_h     = (_Float16*)carve((size_t)QN * HWSZ * DD * 2);
  _Float16* t1       = (_Float16*)carve((size_t)QN * HWSZ * DD * 2);
  _Float16* t2       = (_Float16*)carve((size_t)QN * HWSZ * DD * 2);
  _Float16* w1s_p  = (_Float16*)carve(4 * 1 * 3 * 512 * 2);
  _Float16* w2s_p  = (_Float16*)carve(4 * 1 * 2 * 512 * 2);
  _Float16* w1c_p  = (_Float16*)carve(4 * 1 * 1 * 512 * 2);
  _Float16* w2c_p  = (_Float16*)carve(4 * 1 * 2 * 512 * 2);
  _Float16* hsw1_p = (_Float16*)carve(4 * 9 * 2 * 512 * 2);
  _Float16* hsw2_p = (_Float16*)carve(4 * 9 * 2 * 512 * 2);
  _Float16* hsw3_p = (_Float16*)carve(1 * 9 * 2 * 512 * 2);
  _Float16* how1_p = (_Float16*)carve(4 * 9 * 2 * 512 * 2);
  _Float16* how2_p = (_Float16*)carve(4 * 9 * 2 * 512 * 2);
  _Float16* how3_p = (_Float16*)carve(1 * 9 * 2 * 512 * 2);
  _Float16* hcw1_p = (_Float16*)carve(4 * 9 * 2 * 512 * 2);
  _Float16* hcw2_p = (_Float16*)carve(4 * 9 * 2 * 512 * 2);
  _Float16* hcw3_p = (_Float16*)carve(1 * 9 * 2 * 512 * 2);

  // ---- zero packed-B buffers (xscale_p + xscore_p are contiguous) ----
  int zwords = (int)(((size_t)NTILE * 4 * 512 * 2) / 4);   // 4,194,304 words
  k_zero<<<(zwords + 255) / 256, 256, 0, stream>>>((uint32_t*)xscale_p, zwords);

  // ---- stage 1-4 ----
  k_stats<<<QN * RFN * NSAMP * CN, 256, 0, stream>>>(corr, stats);
  k_lvl<<<24576, 256, 0, stream>>>(corr, stats, lvl);
  k_topk<<<256, 256, 0, stream>>>(lvl, topi);
  k_gather<<<1024, 256, 0, stream>>>(corr, stats, lvl, topi, xscale_p, xscore_p);

  // ---- weight packing into A-fragment layout ----
  auto PACK = [&](const void* src, _Float16* dst, int nmt, int ntaps, int nch,
                  int cout, int cin) {
    int total = nmt * ntaps * nch * 512;
    k_packA<<<(total + 255) / 256, 256, 0, stream>>>((const float*)src, dst,
                                                     nmt, ntaps, nch, cout, cin);
  };
  PACK(d_in[1],  w1s_p,  4, 1, 3, 64, 72);
  PACK(d_in[3],  w2s_p,  4, 1, 2, 64, 64);
  PACK(d_in[5],  w1c_p,  4, 1, 1, 64, 12);
  PACK(d_in[7],  w2c_p,  4, 1, 2, 64, 64);
  PACK(d_in[9],  hsw1_p, 4, 9, 2, 64, 64);
  PACK(d_in[11], hsw2_p, 4, 9, 2, 64, 64);
  PACK(d_in[13], hsw3_p, 1, 9, 2, 1,  64);
  PACK(d_in[15], how1_p, 4, 9, 2, 64, 64);
  PACK(d_in[17], how2_p, 4, 9, 2, 64, 64);
  PACK(d_in[19], how3_p, 1, 9, 2, 2,  64);
  PACK(d_in[21], hcw1_p, 4, 9, 2, 64, 64);
  PACK(d_in[23], hcw2_p, 4, 9, 2, 64, 64);
  PACK(d_in[25], hcw3_p, 1, 9, 2, 1,  64);

  // ---- stage 5-6: 1x1x1 chains + max over s ----
  k_mlp<<<NTILE, 128, 0, stream>>>(xscale_p, 3, w1s_p, (const float*)d_in[2],
                                   w2s_p, (const float*)d_in[4], out2);
  k_maxs<<<4096, 256, 0, stream>>>(out2, xs_h);
  k_mlp<<<NTILE, 128, 0, stream>>>(xscore_p, 1, w1c_p, (const float*)d_in[6],
                                   w2c_p, (const float*)d_in[8], out2);
  k_maxs<<<4096, 256, 0, stream>>>(out2, xc_h);

  // ---- output layout: pos(8) scl(4) scores(16384) scales(16384) offsets(32768)
  float* out     = (float*)d_out;
  float* pos     = out;
  float* scl     = out + 8;
  float* scores  = out + 12;
  float* scales  = out + 12 + 16384;
  float* offsets = out + 12 + 32768;

  // ---- stage 7: heads ----
  k_conv3<<<1024, 128, 0, stream>>>(xs_h, hsw1_p, (const float*)d_in[10], 64, 1, t1, nullptr);
  k_conv3<<<1024, 128, 0, stream>>>(t1,   hsw2_p, (const float*)d_in[12], 64, 1, t2, nullptr);
  k_conv3<<<1024, 32,  0, stream>>>(t2,   hsw3_p, (const float*)d_in[14], 1,  0, nullptr, scales);
  k_conv3<<<1024, 128, 0, stream>>>(xc_h, how1_p, (const float*)d_in[16], 64, 1, t1, nullptr);
  k_conv3<<<1024, 128, 0, stream>>>(t1,   how2_p, (const float*)d_in[18], 64, 1, t2, nullptr);
  k_conv3<<<1024, 32,  0, stream>>>(t2,   how3_p, (const float*)d_in[20], 2,  0, nullptr, offsets);
  k_conv3<<<1024, 128, 0, stream>>>(xc_h, hcw1_p, (const float*)d_in[22], 64, 1, t1, nullptr);
  k_conv3<<<1024, 128, 0, stream>>>(t1,   hcw2_p, (const float*)d_in[24], 64, 1, t2, nullptr);
  k_conv3<<<1024, 32,  0, stream>>>(t2,   hcw3_p, (const float*)d_in[26], 1,  0, nullptr, scores);

  // ---- stage 8 ----
  k_final<<<QN, 256, 0, stream>>>(scores, scales, offsets, pos, scl);
}
